// GATLayer_15384572854698
// MI455X (gfx1250) — compile-verified, matrix operand order checked by reading, not run on hardware
//
#include <hip/hip_runtime.h>

typedef _Float16 h8_t  __attribute__((ext_vector_type(8)));
typedef _Float16 h16_t __attribute__((ext_vector_type(16)));
typedef float    f8_t  __attribute__((ext_vector_type(8)));

constexpr int NB   = 8;     // batch
constexpr int NN   = 1024;  // nodes
constexpr int NFIN = 256;   // in features
constexpr int NH   = 4;     // heads
constexpr int NFO  = 128;   // out features per head
constexpr int NHF  = NH * NFO; // 512

union F16x16 { h16_t v; h8_t h[2]; };

// -------- dynamic LDS layout for k_fused (32-row i-tile) --------
constexpr int    P_PITCH  = 1032;                                   // f16 row pitch, 16B-aligned
constexpr size_t SM_DST   = (size_t)32 * P_PITCH * sizeof(_Float16);// p_s: 66048 B
constexpr size_t SM_SUM   = SM_DST + (size_t)NN * sizeof(float);    // dst_s: 4096 B
constexpr size_t SM_MSK   = SM_SUM + 32 * sizeof(float);            // sum_s: 128 B
constexpr size_t SM_TOTAL = SM_MSK + (size_t)32 * 64 * sizeof(unsigned short); // m_s: 4096 B

// ---------------- convert h (f32) -> f16, row-major [B*N, Fin] ----------------
__global__ __launch_bounds__(256) void k_cvt_h(const float* __restrict__ src,
                                               _Float16* __restrict__ dst) {
  int i = blockIdx.x * 256 + threadIdx.x;
  dst[i] = (_Float16)src[i];
}

// ---------------- convert W (f32 [Fin, H*Fo]) -> Wt f16 [H*Fo, Fin] ----------------
__global__ __launch_bounds__(256) void k_cvt_wT(const float* __restrict__ W,
                                                _Float16* __restrict__ Wt) {
  int i = blockIdx.x * 256 + threadIdx.x;       // i = n*NFIN + k
  int n = i >> 8;
  int k = i & (NFIN - 1);
  Wt[i] = (_Float16)W[k * NHF + n];
}

// ---------------- GEMM1: Wh = h @ W, written transposed as WhT[b][h][f][n] f16 ---
// One wave = one 16x64 C strip: A fragment reused across 4 B fragments (4 WMMAs / 5 loads).
__global__ __launch_bounds__(256) void k_gemm1(const _Float16* __restrict__ A,
                                               const _Float16* __restrict__ Bt,
                                               _Float16* __restrict__ WhT) {
  const int lane = threadIdx.x & 31;
  const int wid  = blockIdx.x * 8 + (threadIdx.x >> 5);
  const int mt   = wid >> 3;             // 512 tiles over rows (b,n)
  const int ns   = wid & 7;              // 8 strips of 64 cols (h,f)
  const int l15  = lane & 15;
  const int hl   = lane >> 4;
  const int kb   = hl * 8;

  const _Float16* ap  = A  + (size_t)(mt * 16 + l15) * NFIN;
  const _Float16* bp0 = Bt + (size_t)(ns * 64 + l15) * NFIN;

  f8_t c[4] = {{}, {}, {}, {}};
  #pragma unroll
  for (int kc = 0; kc < NFIN; kc += 32) {
    F16x16 a;
    a.h[0] = *(const h8_t*)(ap + kc + kb);
    a.h[1] = *(const h8_t*)(ap + kc + 16 + kb);
    #pragma unroll
    for (int u = 0; u < 4; ++u) {
      F16x16 b;
      b.h[0] = *(const h8_t*)(bp0 + (size_t)u * 16 * NFIN + kc + kb);
      b.h[1] = *(const h8_t*)(bp0 + (size_t)u * 16 * NFIN + kc + 16 + kb);
      c[u] = __builtin_amdgcn_wmma_f32_16x16x32_f16(false, a.v, false, b.v,
                                                    (short)0, c[u], false, false);
    }
  }

  // C layout: lane l holds col N=l&15; VGPR v holds row M=v+8*(l>>4).
  const int bb = (mt * 16) >> 10;            // tiles never straddle a batch
  const int n0 = ((mt * 16) & (NN - 1)) + hl * 8;
  #pragma unroll
  for (int u = 0; u < 4; ++u) {
    const int cg = ns * 64 + u * 16 + l15;   // global output col in [0,512)
    const int hh = cg >> 7;
    const int f  = cg & (NFO - 1);
    h8_t st;
    #pragma unroll
    for (int v = 0; v < 8; ++v) st[v] = (_Float16)c[u][v];
    *(h8_t*)(WhT + ((size_t)((bb * NH + hh) * NFO + f)) * NN + n0) = st;
  }
}

// ---------------- attn_src/dst[b,h,n] = <Wh[b,n,h,:], a_src/dst[h,:]> ----------------
__global__ __launch_bounds__(256) void k_attn(const _Float16* __restrict__ WhT,
                                              const float* __restrict__ a_src,
                                              const float* __restrict__ a_dst,
                                              float* __restrict__ sb,
                                              float* __restrict__ db) {
  const int t  = blockIdx.x * 256 + threadIdx.x;  // over NB*NH*NN
  const int n  = t & (NN - 1);
  const int bh = t >> 10;
  const int hh = bh & (NH - 1);
  const _Float16* p = WhT + (size_t)bh * NFO * NN + n;
  float s = 0.f, d = 0.f;
  for (int f = 0; f < NFO; ++f) {
    float v = (float)p[(size_t)f * NN];
    s += v * a_src[hh * NFO + f];
    d += v * a_dst[hh * NFO + f];
  }
  sb[t] = s;
  db[t] = d;
}

// ---------------- fused masked-softmax + attention GEMM + mean + ELU ----------------
// Block = 256 threads (8 waves) = one (b, 32-row i-tile). Heads looped inside.
// Wave w owns output cols f0=w*16; each B fragment feeds TWO WMMAs (rows 0-15 / 16-31),
// halving the dominant L2 WhT stream and giving two independent accumulator chains.
__global__ __launch_bounds__(256) void k_fused(const _Float16* __restrict__ WhT,
                                               const float* __restrict__ asrc,
                                               const float* __restrict__ adst,
                                               const int* __restrict__ adj,
                                               float* __restrict__ out) {
  extern __shared__ char smem[];
  _Float16 (*p_s)[P_PITCH] = (_Float16(*)[P_PITCH])smem;              // [32][1032] f16
  float* dst_s             = (float*)(smem + SM_DST);                 // [1024]
  float* sum_s             = (float*)(smem + SM_SUM);                 // [32]
  unsigned short (*m_s)[64] = (unsigned short(*)[64])(smem + SM_MSK); // [32][64]

  const int t    = threadIdx.x;
  const int bb   = blockIdx.x >> 5;                 // 8 batches x 32 row-groups
  const int i0   = (blockIdx.x & 31) * 32;
  const int lane = t & 31;
  const int w    = t >> 5;
  const int l15  = lane & 15;
  const int hl   = lane >> 4;
  const int r0   = t >> 4;                          // phase-A row 0..15 (+16 for 2nd tile)

  // Pack adjacency once: bit l15 of m_s[r][jj] = (adj[i0+r][l15+16*jj] > 0)
  #pragma unroll
  for (int ht = 0; ht < 2; ++ht) {
    const int r = r0 + 16 * ht;
    const int* arow = adj + (size_t)(i0 + r) * NN;
    for (int jj = 0; jj < 64; ++jj) {
      unsigned long long bal = __ballot(arow[l15 + jj * 16] > 0);
      if (l15 == 0) m_s[r][jj] = (unsigned short)(bal >> (hl * 16));
    }
  }

  float acc0[8] = {0.f, 0.f, 0.f, 0.f, 0.f, 0.f, 0.f, 0.f};
  float acc1[8] = {0.f, 0.f, 0.f, 0.f, 0.f, 0.f, 0.f, 0.f};

  for (int hh = 0; hh < NH; ++hh) {
    __syncthreads();  // previous head's readers of p_s/dst_s done; mask ready (1st iter)
    const float* dsrc = adst + (size_t)(bb * NH + hh) * NN;
    #pragma unroll
    for (int q = t; q < NN; q += 256) dst_s[q] = dsrc[q];
    const float srcv0 = asrc[(size_t)(bb * NH + hh) * NN + i0 + r0];
    const float srcv1 = asrc[(size_t)(bb * NH + hh) * NN + i0 + 16 + r0];
    __syncthreads();

    // Phase A: p[r][j] = mask * exp(leakyrelu(src_r + dst_j)); row sums.
    // exp(-1e9) == 0, so no max-subtraction needed (unmasked logits are O(1)).
    #pragma unroll
    for (int ht = 0; ht < 2; ++ht) {
      const int r = r0 + 16 * ht;
      const float srcv = ht ? srcv1 : srcv0;
      float psum = 0.f;
      for (int jj = 0; jj < 64; ++jj) {
        const int j = l15 + jj * 16;
        float e = srcv + dst_s[j];
        e = e > 0.f ? e : 0.2f * e;
        float p = ((m_s[r][jj] >> l15) & 1) ? __expf(e) : 0.f;
        p_s[r][j] = (_Float16)p;
        psum += p;
      }
      psum += __shfl_xor(psum, 1, 32);
      psum += __shfl_xor(psum, 2, 32);
      psum += __shfl_xor(psum, 4, 32);
      psum += __shfl_xor(psum, 8, 32);
      if (l15 == 0) sum_s[r] = psum;
    }
    __syncthreads();

    // Phase B: C[32x16] = p_tile[32 x 1024] @ Wh[1024 x 16]; row scale 1/sum applied after.
    float inv0[8], inv1[8];
    #pragma unroll
    for (int v = 0; v < 8; ++v) {
      inv0[v] = 1.0f / (sum_s[v + hl * 8] + 1e-30f);
      inv1[v] = 1.0f / (sum_s[16 + v + hl * 8] + 1e-30f);
    }

    const _Float16* bp =
        WhT + ((size_t)((bb * NH + hh) * NFO + w * 16 + l15)) * NN;
    const int kb = hl * 8;
    f8_t c0 = {}, c1 = {};
    #pragma unroll 4
    for (int jc = 0; jc < NN; jc += 32) {
      F16x16 b, a0, a1;
      b.h[0]  = *(const h8_t*)(bp + jc + kb);            // global_load_b128 (L2-resident)
      b.h[1]  = *(const h8_t*)(bp + jc + 16 + kb);
      a0.h[0] = *(const h8_t*)&p_s[l15][jc + kb];        // ds_load_b128
      a0.h[1] = *(const h8_t*)&p_s[l15][jc + 16 + kb];
      a1.h[0] = *(const h8_t*)&p_s[16 + l15][jc + kb];
      a1.h[1] = *(const h8_t*)&p_s[16 + l15][jc + 16 + kb];
      c0 = __builtin_amdgcn_wmma_f32_16x16x32_f16(false, a0.v, false, b.v,
                                                  (short)0, c0, false, false);
      c1 = __builtin_amdgcn_wmma_f32_16x16x32_f16(false, a1.v, false, b.v,
                                                  (short)0, c1, false, false);
    }
    #pragma unroll
    for (int v = 0; v < 8; ++v) {
      acc0[v] += inv0[v] * c0[v];
      acc1[v] += inv1[v] * c1[v];
    }
  }

  // Epilogue: mean over heads + ELU, direct store.
  const int f0 = w * 16;
  #pragma unroll
  for (int v = 0; v < 8; ++v) {
    const int row = hl * 8 + v;
    float x0 = 0.25f * acc0[v];
    x0 = x0 > 0.f ? x0 : (__expf(x0) - 1.f);
    out[((size_t)(bb * NN + i0 + row)) * NFO + f0 + l15] = x0;
    float x1 = 0.25f * acc1[v];
    x1 = x1 > 0.f ? x1 : (__expf(x1) - 1.f);
    out[((size_t)(bb * NN + i0 + 16 + row)) * NFO + f0 + l15] = x1;
  }
}

extern "C" void kernel_launch(void* const* d_in, const int* in_sizes, int n_in,
                              void* d_out, int out_size, void* d_ws, size_t ws_size,
                              hipStream_t stream) {
  (void)in_sizes; (void)n_in; (void)out_size; (void)ws_size;
  const float* h   = (const float*)d_in[0];
  const int*   adj = (const int*)  d_in[1];
  const float* W   = (const float*)d_in[2];
  const float* a_s = (const float*)d_in[3];
  const float* a_d = (const float*)d_in[4];
  float* out = (float*)d_out;

  // Workspace layout (bytes): hf16 4MiB | Wt 256KiB | WhT 8MiB | attn_src 128KiB | attn_dst 128KiB
  char* ws = (char*)d_ws;
  _Float16* hf  = (_Float16*)(ws);
  _Float16* Wt  = (_Float16*)(ws + (size_t)(4u << 20));
  _Float16* WhT = (_Float16*)(ws + (size_t)(4u << 20) + (256u << 10));
  float*    sb  = (float*)(ws + (size_t)(12u << 20) + (256u << 10));
  float*    db  = (float*)(ws + (size_t)(12u << 20) + (256u << 10) + (128u << 10));

  k_cvt_h <<<(NB * NN * NFIN) / 256, 256, 0, stream>>>(h, hf);
  k_cvt_wT<<<(NFIN * NHF) / 256,     256, 0, stream>>>(W, Wt);
  k_gemm1 <<<(NB * NN / 16) * (NHF / 64) / 8, 256, 0, stream>>>(hf, Wt, WhT);
  k_attn  <<<(NB * NH * NN) / 256,   256, 0, stream>>>(WhT, a_s, a_d, sb, db);
  k_fused <<<NB * (NN / 32), 256, (unsigned)SM_TOTAL, stream>>>(WhT, sb, db, adj, out);
}
